// RoIHeads_16887811408561
// MI455X (gfx1250) — matrix operand order repaired and checked by reading, not compiled, not run on hardware
//
#include <hip/hip_runtime.h>
#include <hip/hip_bf16.h>

// ---------------------------------------------------------------------------
// Types for WMMA
// ---------------------------------------------------------------------------
typedef __attribute__((ext_vector_type(16))) __bf16 v16bf;
typedef __attribute__((ext_vector_type(8)))  __bf16 v8bf;
typedef __attribute__((ext_vector_type(4)))  __bf16 v4bf;
typedef __attribute__((ext_vector_type(8)))  float  v8f;

#define R_TOT   4224          // B * (N + G) = 8 * 528
#define PPI     528           // proposals per image (512 + 16)
#define NPROP   512
#define NGT     16
#define NCLS    91
#define FEATD   12544
#define REPD    1024
#define BETA    (1.0f / 9.0f)

struct bfpair { __bf16 hi; __bf16 lo; };

__device__ __forceinline__ bfpair split_f32(float x) {
    bfpair r;
    __bf16 h = (__bf16)x;
    r.hi = h;
    r.lo = (__bf16)(x - (float)h);
    return r;
}

__device__ __forceinline__ v16bf cat8(v8bf a, v8bf b) {
    v16bf r;
#pragma unroll
    for (int i = 0; i < 8; ++i) { r[i] = a[i]; r[i + 8] = b[i]; }
    return r;
}

// ---------------------------------------------------------------------------
// Kernel 1: proposal-GT matching + box target encoding
// ---------------------------------------------------------------------------
__global__ void assign_kernel(const float* __restrict__ proposals,
                              const float* __restrict__ gt_boxes,
                              const int*   __restrict__ gt_labels,
                              int*   __restrict__ labels,
                              float* __restrict__ tgt)
{
    int r = blockIdx.x * blockDim.x + threadIdx.x;
    if (r >= R_TOT) return;
    int b = r / PPI, p = r % PPI;
    const float* pr = (p < NPROP)
        ? proposals + ((size_t)b * NPROP + p) * 4
        : gt_boxes  + ((size_t)b * NGT + (p - NPROP)) * 4;
    float px0 = pr[0], py0 = pr[1], px1 = pr[2], py1 = pr[3];
    float pa = (px1 - px0) * (py1 - py0);

    float best = -1.0f; int bi = 0;
#pragma unroll
    for (int g = 0; g < NGT; ++g) {
        const float* gb = gt_boxes + ((size_t)b * NGT + g) * 4;
        float gx0 = gb[0], gy0 = gb[1], gx1 = gb[2], gy1 = gb[3];
        float ga = (gx1 - gx0) * (gy1 - gy0);
        float ix0 = fmaxf(gx0, px0), iy0 = fmaxf(gy0, py0);
        float ix1 = fminf(gx1, px1), iy1 = fminf(gy1, py1);
        float iw = fmaxf(ix1 - ix0, 0.0f), ih = fmaxf(iy1 - iy0, 0.0f);
        float inter = iw * ih;
        float iou = inter / (ga + pa - inter);
        if (iou > best) { best = iou; bi = g; }   // strict > keeps first argmax
    }
    int lab = gt_labels[b * NGT + bi];
    if (best < 0.5f)       lab = 0;    // background (BG threshold)
    else if (best < 0.5f)  lab = -1;   // ignore band (empty since FG==BG)

    const float* gb = gt_boxes + ((size_t)b * NGT + bi) * 4;
    float ew = px1 - px0, eh = py1 - py0;
    float ex = px0 + 0.5f * ew, ey = py0 + 0.5f * eh;
    float gw = gb[2] - gb[0], gh = gb[3] - gb[1];
    float gx = gb[0] + 0.5f * gw, gy = gb[1] + 0.5f * gh;
    tgt[4 * r + 0] = 10.0f * (gx - ex) / ew;
    tgt[4 * r + 1] = 10.0f * (gy - ey) / eh;
    tgt[4 * r + 2] = 5.0f * logf(gw / ew);
    tgt[4 * r + 3] = 5.0f * logf(gh / eh);
    labels[r] = lab;
}

// ---------------------------------------------------------------------------
// Kernel 2: generic GEMM  C = act(A[M,K] @ W[K,N] + bias)
// bf16x3 split-precision via v_wmma_f32_16x16x32_bf16.
// Block = 128x128 output, BK=32 per stage, 256 threads (8 waves),
// wave -> 32x64 region = 2x4 tiles of 16x16.
// Register-prefetch pipeline: global loads of tile k+1 are issued before the
// WMMA burst for tile k, so HBM latency hides under matrix math (LOADcnt is
// only waited at the next stage's convert/store).
// M % 128 == 0, K % 32 == 0 required; N arbitrary (guarded).
// ---------------------------------------------------------------------------
__global__ __launch_bounds__(256) void gemm_bf16x3(
    const float* __restrict__ A, const float* __restrict__ Bw,
    const float* __restrict__ bias, float* __restrict__ Cout,
    int M, int N, int K, int doRelu)
{
    // row stride 40 bf16 = 80B: multiple of 16B, conflict-free b128 frag loads
    __shared__ alignas(16) __bf16 As_hi[128][40];
    __shared__ alignas(16) __bf16 As_lo[128][40];
    __shared__ alignas(16) __bf16 Bs_hi[128][40];   // [col][k] (K-contiguous)
    __shared__ alignas(16) __bf16 Bs_lo[128][40];

    const int t     = threadIdx.x;
    const int wave  = t >> 5;
    const int lane  = t & 31;
    const int lm    = lane & 15;
    const int lh    = lane >> 4;
    const int waveM = wave & 3;     // rows  32*waveM .. +32
    const int waveN = wave >> 2;    // cols  64*waveN .. +64
    const int mBase = blockIdx.y * 128;
    const int nBase = blockIdx.x * 128;

    // loop-invariant staging coordinates
    const int aRow = t >> 3;          // 0..31   (+32*j)
    const int aK4  = (t & 7) << 2;    // 0,4,...,28
    const int bKr  = t >> 5;          // 0..7    (+8*j)
    const int bC4  = (t & 31) << 2;   // 0,4,...,124

    float4 aReg[4];
    float  bReg[4][4];

    auto issue_loads = [&](int ks) {
#pragma unroll
        for (int j = 0; j < 4; ++j) {
            int row = aRow + 32 * j;
            aReg[j] = *(const float4*)(A + (size_t)(mBase + row) * K + ks + aK4);
        }
#pragma unroll
        for (int j = 0; j < 4; ++j) {
            int kr = bKr + 8 * j;
#pragma unroll
            for (int i = 0; i < 4; ++i) {
                int n = nBase + bC4 + i;
                bReg[j][i] = (n < N) ? Bw[(size_t)(ks + kr) * N + n] : 0.0f;
            }
        }
    };

    auto stage_to_lds = [&]() {
#pragma unroll
        for (int j = 0; j < 4; ++j) {
            int row = aRow + 32 * j;
            float xs[4] = {aReg[j].x, aReg[j].y, aReg[j].z, aReg[j].w};
            v4bf hv, lv;
#pragma unroll
            for (int i = 0; i < 4; ++i) {
                bfpair s = split_f32(xs[i]);
                hv[i] = s.hi;
                lv[i] = s.lo;
            }
            *(v4bf*)&As_hi[row][aK4] = hv;
            *(v4bf*)&As_lo[row][aK4] = lv;
        }
#pragma unroll
        for (int j = 0; j < 4; ++j) {
            int kr = bKr + 8 * j;
#pragma unroll
            for (int i = 0; i < 4; ++i) {
                bfpair s = split_f32(bReg[j][i]);
                Bs_hi[bC4 + i][kr] = s.hi;
                Bs_lo[bC4 + i][kr] = s.lo;
            }
        }
    };

    v8f zero;
#pragma unroll
    for (int i = 0; i < 8; ++i) zero[i] = 0.0f;
    v8f acc[2][4];
#pragma unroll
    for (int rt = 0; rt < 2; ++rt)
#pragma unroll
        for (int ct = 0; ct < 4; ++ct) acc[rt][ct] = zero;

    const int kTiles = K >> 5;
    issue_loads(0);                            // prologue

    for (int kt = 0; kt < kTiles; ++kt) {
        stage_to_lds();                        // waits LOADcnt, converts, DS stores
        __syncthreads();
        if (kt + 1 < kTiles)
            issue_loads((kt + 1) << 5);        // in-flight during WMMA burst

        // ---- A fragments (16-bit A 16x32 layout) ----
        v16bf ahi[2], alo[2];
#pragma unroll
        for (int rt = 0; rt < 2; ++rt) {
            int row = waveM * 32 + rt * 16 + lm;
            ahi[rt] = cat8(*(const v8bf*)&As_hi[row][lh * 16],
                           *(const v8bf*)&As_hi[row][lh * 16 + 8]);
            alo[rt] = cat8(*(const v8bf*)&As_lo[row][lh * 16],
                           *(const v8bf*)&As_lo[row][lh * 16 + 8]);
        }
        // ---- B fragments (16-bit B 32x16 layout) + WMMA ----
#pragma unroll
        for (int ct = 0; ct < 4; ++ct) {
            int col = waveN * 64 + ct * 16 + lm;
            v16bf bhi = cat8(*(const v8bf*)&Bs_hi[col][lh * 16],
                             *(const v8bf*)&Bs_hi[col][lh * 16 + 8]);
            v16bf blo = cat8(*(const v8bf*)&Bs_lo[col][lh * 16],
                             *(const v8bf*)&Bs_lo[col][lh * 16 + 8]);
#pragma unroll
            for (int rt = 0; rt < 2; ++rt) {
                acc[rt][ct] = __builtin_amdgcn_wmma_f32_16x16x32_bf16(
                    false, ahi[rt], false, bhi, (short)0, acc[rt][ct], false, false);
                acc[rt][ct] = __builtin_amdgcn_wmma_f32_16x16x32_bf16(
                    false, ahi[rt], false, blo, (short)0, acc[rt][ct], false, false);
                acc[rt][ct] = __builtin_amdgcn_wmma_f32_16x16x32_bf16(
                    false, alo[rt], false, bhi, (short)0, acc[rt][ct], false, false);
            }
        }
        __syncthreads();
    }

    // ---- epilogue: C/D 16x16 layout (VGPR r: M=r+8*lh, N=lm) ----
#pragma unroll
    for (int rt = 0; rt < 2; ++rt) {
#pragma unroll
        for (int ct = 0; ct < 4; ++ct) {
            int col = nBase + waveN * 64 + ct * 16 + lm;
            if (col < N) {
                float bv = bias[col];
#pragma unroll
                for (int r = 0; r < 8; ++r) {
                    int row = mBase + waveM * 32 + rt * 16 + r + lh * 8;
                    float v = acc[rt][ct][r] + bv;
                    if (doRelu) v = fmaxf(v, 0.0f);
                    Cout[(size_t)row * N + col] = v;
                }
            }
        }
    }
}

// ---------------------------------------------------------------------------
// Kernel 3: per-row CE + smooth-L1 terms
// ---------------------------------------------------------------------------
__global__ void rowloss_kernel(const float* __restrict__ cls,
                               const float* __restrict__ box,
                               const int*   __restrict__ labels,
                               const float* __restrict__ tgt,
                               float* __restrict__ cterm,
                               float* __restrict__ vterm,
                               float* __restrict__ bterm)
{
    int r = blockIdx.x * blockDim.x + threadIdx.x;
    if (r >= R_TOT) return;
    const float* lg = cls + (size_t)r * NCLS;
    float mx = lg[0];
    for (int i = 1; i < NCLS; ++i) mx = fmaxf(mx, lg[i]);
    float s = 0.0f;
    for (int i = 0; i < NCLS; ++i) s += expf(lg[i] - mx);
    float lse = mx + logf(s);

    int lab = labels[r];
    int sl  = lab < 0 ? 0 : (lab > NCLS - 1 ? NCLS - 1 : lab);
    float valid = (lab >= 0) ? 1.0f : 0.0f;
    cterm[r] = (lse - lg[sl]) * valid;
    vterm[r] = valid;

    const float* pd = box + (size_t)r * (NCLS * 4) + sl * 4;
    const float* tg = tgt + (size_t)r * 4;
    float sum = 0.0f;
#pragma unroll
    for (int i = 0; i < 4; ++i) {
        float d = fabsf(pd[i] - tg[i]);
        sum += (d < BETA) ? 0.5f * d * d / BETA : d - 0.5f * BETA;
    }
    bterm[r] = (lab > 0) ? sum : 0.0f;
}

// ---------------------------------------------------------------------------
// Kernel 4: deterministic single-block reduction -> d_out[0..1]
// ---------------------------------------------------------------------------
__global__ void reduce_kernel(const float* __restrict__ cterm,
                              const float* __restrict__ vterm,
                              const float* __restrict__ bterm,
                              float* __restrict__ out)
{
    __shared__ float s0[256], s1[256], s2[256];
    int t = threadIdx.x;
    float a = 0.0f, b = 0.0f, c = 0.0f;
    for (int i = t; i < R_TOT; i += 256) { a += cterm[i]; b += vterm[i]; c += bterm[i]; }
    s0[t] = a; s1[t] = b; s2[t] = c;
    __syncthreads();
    for (int off = 128; off > 0; off >>= 1) {
        if (t < off) { s0[t] += s0[t + off]; s1[t] += s1[t + off]; s2[t] += s2[t + off]; }
        __syncthreads();
    }
    if (t == 0) {
        out[0] = s0[0] / fmaxf(s1[0], 1.0f);
        out[1] = s2[0] / (float)R_TOT;
    }
}

// ---------------------------------------------------------------------------
// Launch
// ---------------------------------------------------------------------------
extern "C" void kernel_launch(void* const* d_in, const int* in_sizes, int n_in,
                              void* d_out, int out_size, void* d_ws, size_t ws_size,
                              hipStream_t stream) {
    const float* proposals = (const float*)d_in[0];
    const float* gt_boxes  = (const float*)d_in[1];
    const float* features  = (const float*)d_in[2];
    const float* W1   = (const float*)d_in[3];
    const float* b1   = (const float*)d_in[4];
    const float* W2   = (const float*)d_in[5];
    const float* b2   = (const float*)d_in[6];
    const float* Wcls = (const float*)d_in[7];
    const float* bcls = (const float*)d_in[8];
    const float* Wbox = (const float*)d_in[9];
    const float* bbox = (const float*)d_in[10];
    const int*   gt_labels = (const int*)d_in[11];

    float* ws = (float*)d_ws;
    size_t off = 0;
    int*   labels = (int*)(ws + off);            off += R_TOT;
    float* tgt    = ws + off;                    off += (size_t)R_TOT * 4;
    float* x1     = ws + off;                    off += (size_t)R_TOT * REPD;
    float* x2     = ws + off;                    off += (size_t)R_TOT * REPD;
    float* clsb   = ws + off;                    off += (size_t)R_TOT * NCLS;
    float* boxb   = ws + off;                    off += (size_t)R_TOT * NCLS * 4;
    float* cterm  = ws + off;                    off += R_TOT;
    float* vterm  = ws + off;                    off += R_TOT;
    float* bterm  = ws + off;                    off += R_TOT;

    assign_kernel<<<(R_TOT + 255) / 256, 256, 0, stream>>>(
        proposals, gt_boxes, gt_labels, labels, tgt);

    // x1 = relu(features @ W1 + b1)   [4224,12544]x[12544,1024]
    gemm_bf16x3<<<dim3(REPD / 128, R_TOT / 128), 256, 0, stream>>>(
        features, W1, b1, x1, R_TOT, REPD, FEATD, 1);
    // x2 = relu(x1 @ W2 + b2)         [4224,1024]x[1024,1024]
    gemm_bf16x3<<<dim3(REPD / 128, R_TOT / 128), 256, 0, stream>>>(
        x1, W2, b2, x2, R_TOT, REPD, REPD, 1);
    // class_logits = x2 @ Wcls + bcls  (N=91)
    gemm_bf16x3<<<dim3(1, R_TOT / 128), 256, 0, stream>>>(
        x2, Wcls, bcls, clsb, R_TOT, NCLS, REPD, 0);
    // box_regression = x2 @ Wbox + bbox  (N=364)
    gemm_bf16x3<<<dim3((NCLS * 4 + 127) / 128, R_TOT / 128), 256, 0, stream>>>(
        x2, Wbox, bbox, boxb, R_TOT, NCLS * 4, REPD, 0);

    rowloss_kernel<<<(R_TOT + 255) / 256, 256, 0, stream>>>(
        clsb, boxb, labels, tgt, cterm, vterm, bterm);
    reduce_kernel<<<1, 256, 0, stream>>>(cterm, vterm, bterm, (float*)d_out);
}
